// Block_89859305767239
// MI455X (gfx1250) — matrix operand order repaired and checked by reading, not compiled, not run on hardware
//
#include <hip/hip_runtime.h>
#include <hip/hip_bf16.h>

// ---------- types ----------
typedef __bf16 bf16;
typedef __attribute__((ext_vector_type(16))) __bf16 bf16x16;
typedef __attribute__((ext_vector_type(8)))  __bf16 bf16x8;
typedef __attribute__((ext_vector_type(8)))  float  f32x8;

#define WMMA_BF16(a, b, c) \
  __builtin_amdgcn_wmma_f32_16x16x32_bf16(false, (a), false, (b), (short)0, (c), false, false)

__device__ __forceinline__ bf16x16 join8(bf16x8 lo, bf16x8 hi) {
  bf16x16 r;
#pragma unroll
  for (int i = 0; i < 8; ++i) { r[i] = lo[i]; r[i + 8] = hi[i]; }
  return r;
}

__device__ __forceinline__ float rowmax16(float v) {
  v = fmaxf(v, __shfl_xor(v, 1));
  v = fmaxf(v, __shfl_xor(v, 2));
  v = fmaxf(v, __shfl_xor(v, 4));
  v = fmaxf(v, __shfl_xor(v, 8));
  return v;
}
__device__ __forceinline__ float rowsum16(float v) {
  v += __shfl_xor(v, 1);
  v += __shfl_xor(v, 2);
  v += __shfl_xor(v, 4);
  v += __shfl_xor(v, 8);
  return v;
}

// ---------- fp32 -> bf16 bulk convert (n multiple of 8) ----------
__global__ void cvt_f32_bf16(const float* __restrict__ src, bf16* __restrict__ dst, int n) {
  const int i = (blockIdx.x * blockDim.x + threadIdx.x) * 8;
  if (i >= n) return;
  const float4* s = (const float4*)(src + i);
  const float4 a = s[0], b = s[1];
  bf16x8 o;
  o[0] = (bf16)a.x; o[1] = (bf16)a.y; o[2] = (bf16)a.z; o[3] = (bf16)a.w;
  o[4] = (bf16)b.x; o[5] = (bf16)b.y; o[6] = (bf16)b.z; o[7] = (bf16)b.w;
  *(bf16x8*)(dst + i) = o;
}

// ---------- LayerNorm: fp32 in -> bf16 normalized out (D == 768) ----------
__global__ void ln_kernel(const float* __restrict__ x, bf16* __restrict__ out, int D) {
  const int row = blockIdx.x;
  const int tid = threadIdx.x;
  const float* xr = x + (size_t)row * D;

  float vals[3];
  float s = 0.f, sq = 0.f;
#pragma unroll
  for (int j = 0; j < 3; ++j) {
    float v = xr[tid + j * 256];
    vals[j] = v;
    s += v;
    sq += v * v;
  }
  __shared__ float sh1[256], sh2[256];
  sh1[tid] = s; sh2[tid] = sq;
  __syncthreads();
  for (int st = 128; st > 0; st >>= 1) {
    if (tid < st) { sh1[tid] += sh1[tid + st]; sh2[tid] += sh2[tid + st]; }
    __syncthreads();
  }
  const float mean = sh1[0] / (float)D;
  const float var  = (sh2[0] - (float)D * mean * mean) / (float)(D - 1);  // unbiased (ddof=1)
  const float rstd = rsqrtf(var + 1e-5f);
#pragma unroll
  for (int j = 0; j < 3; ++j)
    out[(size_t)row * D + tid + j * 256] = (bf16)((vals[j] - mean) * rstd);
}

// ---------- WMMA GEMM, 16x64 per wave: out[m,n] = sum_k A[m,k]*W[n,k] (+bias +res, ReLU) ----------
// A: bf16 [M,K] row-major. Wb: bf16 [N,K] (pre-converted torch Linear layout).
// outF (fp32 [M,N]) and/or outB (bf16; normal [M,N] or transposed [N,M] if transB).
__global__ void gemm_wmma(const bf16* __restrict__ A, const bf16* __restrict__ Wb,
                          const float* __restrict__ bias, const float* __restrict__ res,
                          bf16* __restrict__ outB, float* __restrict__ outF,
                          int M, int N, int K, int relu, int transB) {
  const int gtid = blockIdx.x * blockDim.x + threadIdx.x;
  const int wave = gtid >> 5;
  const int lane = threadIdx.x & 31;
  const int nCols = N >> 6;                 // 64-wide N blocks
  const int tiles = (M >> 4) * nCols;
  if (wave >= tiles) return;

  const int tm = wave / nCols, tn = wave % nCols;
  const int m0 = tm << 4, n0 = tn << 6;
  const int half = lane >> 4, nl = lane & 15;

  f32x8 acc[4] = {{}, {}, {}, {}};
  const bf16* arow = A + (size_t)(m0 + nl) * K;
  const bf16* wr   = Wb + (size_t)(n0 + nl) * K;   // n-tile stride = 16*K

  for (int k0 = 0; k0 < K; k0 += 32) {
    // A fragment: element e -> k = k0 + 16*half + e (16 contiguous bf16)
    bf16x16 a = *(const bf16x16*)(arow + k0 + 16 * half);
#pragma unroll
    for (int nt = 0; nt < 4; ++nt) {
      // B fragment: element e -> k = k0 + 16*half + e (16 contiguous bf16)
      bf16x16 b = *(const bf16x16*)(wr + (size_t)nt * 16 * K + k0 + 16 * half);
      acc[nt] = WMMA_BF16(a, b, acc[nt]);
    }
  }

#pragma unroll
  for (int nt = 0; nt < 4; ++nt) {
    const int n = n0 + nt * 16 + nl;
    const float bval = bias ? bias[n] : 0.f;
#pragma unroll
    for (int r = 0; r < 8; ++r) {
      const int m = m0 + r + 8 * half;
      float v = acc[nt][r] + bval;
      if (relu) v = fmaxf(v, 0.f);
      if (res)  v += res[(size_t)m * N + n];
      if (outF) outF[(size_t)m * N + n] = v;
      if (outB) {
        if (transB) outB[(size_t)n * M + m] = (bf16)v;
        else        outB[(size_t)m * N + n] = (bf16)v;
      }
    }
  }
}

// ---------- Attention: S = K·Q^T * scale, causal softmax, attn -> d_out, o = attn·V ----------
// q,k: bf16 [B*T, 768] (col = h*64+d). vT: bf16 [768, B*T] (row = h*64+d, col = b*T+s).
// attnOut: fp32 [B,H,T,T]. oOut: bf16 [B*T, 768].
__global__ void attn_kernel(const bf16* __restrict__ q, const bf16* __restrict__ k,
                            const bf16* __restrict__ vT, float* __restrict__ attnOut,
                            bf16* __restrict__ oOut, int B, int H, int T) {
  __shared__ float lds[4][2][16][16];
  const int wv = threadIdx.x >> 5;
  const int lane = threadIdx.x & 31;
  const int gw = blockIdx.x * 4 + wv;
  const int tilesT = T >> 4;                 // 128
  const int tTile = gw % tilesT;
  const int bh = gw / tilesT;
  const int h = bh % H, b = bh / H;
  if (b >= B) return;

  const int D = 768, HS = 64;
  const int half = lane >> 4, nl = lane & 15;
  const int t0 = tTile << 4;
  const size_t BT = (size_t)B * T;
  const float scale = 0.03608439182435161f;  // 768^-0.5 (d_model scaling per reference)

  const bf16* qB = q  + (size_t)b * T * D + h * HS;
  const bf16* kB = k  + (size_t)b * T * D + h * HS;
  const bf16* vB = vT + (size_t)(h * HS) * BT + (size_t)b * T;
  float* aOut = attnOut + ((size_t)(b * H + h)) * T * T;

  // K A-fragments are invariant over the s loop.
  const bf16* krow = kB + (size_t)(t0 + nl) * D;
  bf16x16 ka0 = *(const bf16x16*)(krow + 16 * half);
  bf16x16 ka1 = *(const bf16x16*)(krow + 32 + 16 * half);

  float mrow[8], lrow[8];
#pragma unroll
  for (int r = 0; r < 8; ++r) { mrow[r] = -3.0e38f; lrow[r] = 0.f; }

  // ---- pass 1: online row max / sum over causal tiles ----
  for (int st = 0; st <= tTile; ++st) {
    const int s0 = st << 4;
    const bf16* qrow = qB + (size_t)(s0 + nl) * D;
    bf16x16 qb0 = *(const bf16x16*)(qrow + 16 * half);
    bf16x16 qb1 = *(const bf16x16*)(qrow + 32 + 16 * half);
    f32x8 S = {};
    S = WMMA_BF16(ka0, qb0, S);
    S = WMMA_BF16(ka1, qb1, S);
#pragma unroll
    for (int r = 0; r < 8; ++r) {
      float sv = S[r] * scale;
      if (st == tTile) {
        const int s = s0 + nl, t = t0 + r + 8 * half;
        if (s > t) sv = -3.0e38f;
      }
      const float tmax = rowmax16(sv);
      const float nm = fmaxf(mrow[r], tmax);
      const float p = rowsum16(__expf(sv - nm));
      lrow[r] = lrow[r] * __expf(mrow[r] - nm) + p;
      mrow[r] = nm;
    }
  }
  float invl[8];
#pragma unroll
  for (int r = 0; r < 8; ++r) invl[r] = 1.f / lrow[r];

  // ---- pass 2: write attn, accumulate o = attn @ V ----
  f32x8 oacc[4] = {{}, {}, {}, {}};
  for (int stp = 0; stp < tilesT; stp += 2) {
#pragma unroll
    for (int sub = 0; sub < 2; ++sub) {
      const int st = stp + sub, s0 = st << 4;
      float av[8];
      if (st <= tTile) {
        const bf16* qrow = qB + (size_t)(s0 + nl) * D;
        bf16x16 qb0 = *(const bf16x16*)(qrow + 16 * half);
        bf16x16 qb1 = *(const bf16x16*)(qrow + 32 + 16 * half);
        f32x8 S = {};
        S = WMMA_BF16(ka0, qb0, S);
        S = WMMA_BF16(ka1, qb1, S);
#pragma unroll
        for (int r = 0; r < 8; ++r) {
          const float sv = S[r] * scale;
          const int s = s0 + nl, t = t0 + r + 8 * half;
          const bool msk = (st == tTile) && (s > t);
          av[r] = msk ? 0.f : __expf(sv - mrow[r]) * invl[r];
        }
      } else {
#pragma unroll
        for (int r = 0; r < 8; ++r) av[r] = 0.f;
      }
      // global store (row-coalesced across the 16 lanes of each half) + LDS for transpose
#pragma unroll
      for (int r = 0; r < 8; ++r) {
        const int t = t0 + r + 8 * half;
        aOut[(size_t)t * T + s0 + nl] = av[r];
        lds[wv][sub][r + 8 * half][nl] = av[r];
      }
    }
    if (stp <= tTile) {
      // attn tile-pair as A fragment (transpose via LDS):
      //   element e: s = 16*half + e  (sub = half)
      bf16x16 aa;
#pragma unroll
      for (int j = 0; j < 8; ++j) {
        aa[j]     = (bf16)lds[wv][half][nl][j];
        aa[j + 8] = (bf16)lds[wv][half][nl][8 + j];
      }
      const int sBase = stp << 4;
#pragma unroll
      for (int nt = 0; nt < 4; ++nt) {
        // V B-fragment: element e -> K(=s_local) = 16*half + e, N(=d) = nt*16 + nl
        const bf16* vrow = vB + (size_t)(nt * 16 + nl) * BT + sBase + 16 * half;
        bf16x16 vb = *(const bf16x16*)vrow;
        oacc[nt] = WMMA_BF16(aa, vb, oacc[nt]);
      }
    }
  }
  // write o (bf16 [B*T, 768], col = h*64 + d)
#pragma unroll
  for (int nt = 0; nt < 4; ++nt)
#pragma unroll
    for (int r = 0; r < 8; ++r)
      oOut[(size_t)((size_t)b * T + t0 + r + 8 * half) * 768 + h * 64 + nt * 16 + nl] =
          (bf16)oacc[nt][r];
}

// ---------- host orchestration ----------
extern "C" void kernel_launch(void* const* d_in, const int* in_sizes, int n_in,
                              void* d_out, int out_size, void* d_ws, size_t ws_size,
                              hipStream_t stream) {
  (void)in_sizes; (void)n_in; (void)out_size; (void)ws_size;
  const int B = 2, T = 2048, Dm = 768, H = 12, HID = 3072;
  const int M = B * T;  // 4096 rows

  const float* x  = (const float*)d_in[0];
  const float* Wq = (const float*)d_in[1];
  const float* bq = (const float*)d_in[2];
  const float* Wk = (const float*)d_in[3];
  const float* bk = (const float*)d_in[4];
  const float* Wv = (const float*)d_in[5];
  const float* bv = (const float*)d_in[6];
  const float* Wp = (const float*)d_in[7];
  const float* bp = (const float*)d_in[8];
  const float* W1 = (const float*)d_in[9];
  const float* b1 = (const float*)d_in[10];
  const float* W2 = (const float*)d_in[11];
  const float* b2 = (const float*)d_in[12];

  float* outX    = (float*)d_out;
  float* outAttn = outX + (size_t)M * Dm;  // attn [B,H,T,T] follows x [B,T,D]

  char* ws = (char*)d_ws;
  size_t off = 0;
  auto alloc = [&](size_t bytes) -> void* {
    void* p = ws + off;
    off += (bytes + 255) & ~(size_t)255;
    return p;
  };
  bf16*  xn  = (bf16*) alloc((size_t)M * Dm * 2);
  bf16*  qw  = (bf16*) alloc((size_t)M * Dm * 2);
  bf16*  kw  = (bf16*) alloc((size_t)M * Dm * 2);
  bf16*  vTw = (bf16*) alloc((size_t)M * Dm * 2);   // [768, 4096]
  bf16*  ow  = (bf16*) alloc((size_t)M * Dm * 2);
  float* x2  = (float*)alloc((size_t)M * Dm * 4);
  bf16*  xn2 = (bf16*) alloc((size_t)M * Dm * 2);
  bf16*  hw  = (bf16*) alloc((size_t)M * HID * 2);
  // bf16 weight copies
  const int nQ = Dm * Dm;       // 589824
  const int nM1 = HID * Dm;     // 2359296
  bf16* Wqb = (bf16*)alloc((size_t)nQ * 2);
  bf16* Wkb = (bf16*)alloc((size_t)nQ * 2);
  bf16* Wvb = (bf16*)alloc((size_t)nQ * 2);
  bf16* Wpb = (bf16*)alloc((size_t)nQ * 2);
  bf16* W1b = (bf16*)alloc((size_t)nM1 * 2);
  bf16* W2b = (bf16*)alloc((size_t)nM1 * 2);

  auto cvt_blocks  = [](int n) { return (n / 8 + 255) / 256; };
  auto gemm_blocks = [](int m, int n) { return (((m >> 4) * (n >> 6)) + 7) / 8; };

  // 0) one-time weight conversion fp32 -> bf16
  cvt_f32_bf16<<<cvt_blocks(nQ),  256, 0, stream>>>(Wq, Wqb, nQ);
  cvt_f32_bf16<<<cvt_blocks(nQ),  256, 0, stream>>>(Wk, Wkb, nQ);
  cvt_f32_bf16<<<cvt_blocks(nQ),  256, 0, stream>>>(Wv, Wvb, nQ);
  cvt_f32_bf16<<<cvt_blocks(nQ),  256, 0, stream>>>(Wp, Wpb, nQ);
  cvt_f32_bf16<<<cvt_blocks(nM1), 256, 0, stream>>>(W1, W1b, nM1);
  cvt_f32_bf16<<<cvt_blocks(nM1), 256, 0, stream>>>(W2, W2b, nM1);

  // 1) LN1
  ln_kernel<<<M, 256, 0, stream>>>(x, xn, Dm);
  // 2-4) Q, K, V projections (V stored transposed)
  gemm_wmma<<<gemm_blocks(M, Dm), 256, 0, stream>>>(xn, Wqb, bq, nullptr, qw,  nullptr, M, Dm, Dm, 0, 0);
  gemm_wmma<<<gemm_blocks(M, Dm), 256, 0, stream>>>(xn, Wkb, bk, nullptr, kw,  nullptr, M, Dm, Dm, 0, 0);
  gemm_wmma<<<gemm_blocks(M, Dm), 256, 0, stream>>>(xn, Wvb, bv, nullptr, vTw, nullptr, M, Dm, Dm, 0, 1);
  // 5) attention (writes attn probabilities to d_out, o to ws)
  {
    const int waves = B * H * (T >> 4);   // 3072
    attn_kernel<<<waves / 4, 128, 0, stream>>>(qw, kw, vTw, outAttn, ow, B, H, T);
  }
  // 6) proj + residual: x2 = x + o @ Wp^T + bp
  gemm_wmma<<<gemm_blocks(M, Dm), 256, 0, stream>>>(ow, Wpb, bp, x, nullptr, x2, M, Dm, Dm, 0, 0);
  // 7) LN2
  ln_kernel<<<M, 256, 0, stream>>>(x2, xn2, Dm);
  // 8) MLP1: h = relu(xn2 @ W1^T + b1)
  gemm_wmma<<<gemm_blocks(M, HID), 256, 0, stream>>>(xn2, W1b, b1, nullptr, hw, nullptr, M, HID, Dm, 1, 0);
  // 9) MLP2 + residual: out = x2 + h @ W2^T + b2
  gemm_wmma<<<gemm_blocks(M, Dm), 256, 0, stream>>>(hw, W2b, b2, x2, nullptr, outX, M, Dm, HID, 0, 0);
}
